// RModel_70308614636142
// MI455X (gfx1250) — matrix operand order repaired and checked by reading, not compiled, not run on hardware
//
#include <hip/hip_runtime.h>
#include <hip/hip_bf16.h>

typedef __attribute__((ext_vector_type(16))) _Float16 v16h;
typedef __attribute__((ext_vector_type(8)))  float    v8f;

#define NPIX    (16 * 256 * 256)
#define NGROUPS (NPIX / 16)

// SWAPX16 across the wave32: group-of-32 swizzle, xor=0x10, or=0, and=0x1f
__device__ __forceinline__ float xor16(float v) {
    int i = __float_as_int(v);
    i = __builtin_amdgcn_ds_swizzle(i, 0x401f);
    return __int_as_float(i);
}

// cos via hardware V_COS_F32 (revolutions input) with one Cody-Waite
// reduction step. Valid/accurate for |x| up to ~1e4; our args are < ~300.
__device__ __forceinline__ float fast_cosf(float x) {
    const float INV2PI  = 0.15915494309189535f;   // 1/(2*pi)
    const float TWOPI_H = 6.2831854820251465f;    // float(2*pi)
    const float TWOPI_L = 1.7484555314695172e-7f; // TWOPI_H - 2*pi (exact tail)
    float n = rintf(x * INV2PI);
    float r = fmaf(n, -TWOPI_H, x);   // r = x - n*2pi (hi part)
    r = fmaf(n, -TWOPI_L, r);         // fold in the tail
    return __builtin_amdgcn_cosf(r * INV2PI);  // v_cos_f32, input in revolutions
}

__global__ __launch_bounds__(256) void siren_warp_wmma(
    const float* __restrict__ xg,     // (NPIX, 3)
    const float* __restrict__ gen,    // (2,2)
    const float* __restrict__ trg,    // (2,)
    const float* __restrict__ w0g,    // (2,32)
    const float* __restrict__ b0g,    // (32,)
    const float* __restrict__ w1g,    // (32,32)
    const float* __restrict__ b1g,    // (32,)
    const float* __restrict__ w2g,    // (32,32)
    const float* __restrict__ b2g,    // (32,)
    const float* __restrict__ gmg,    // bn_gamma (32,)
    const float* __restrict__ btg,    // bn_beta  (32,)
    const float* __restrict__ mng,    // bn_mean  (32,)
    const float* __restrict__ vrg,    // bn_var   (32,)
    const float* __restrict__ wfg,    // (32,2)
    const float* __restrict__ img,    // (512,512,1)
    float* __restrict__ out)          // (NPIX,)
{
    const int lane = threadIdx.x & 31;
    const int p    = lane & 15;   // pixel-in-group / A-matrix row M
    const int hi   = lane >> 4;   // half-wave select

    // ---------------- expm(gen), 2x2 float Taylor (||gen|| ~ 0.02) ----------
    float g00 = gen[0], g01 = gen[1], g10 = gen[2], g11 = gen[3];
    float a00 = 1.f, a01 = 0.f, a10 = 0.f, a11 = 1.f;
    float t00 = 1.f, t01 = 0.f, t10 = 0.f, t11 = 1.f;
    #pragma unroll
    for (int n = 1; n <= 12; ++n) {
        float inv = 1.0f / (float)n;
        float u00 = (t00 * g00 + t01 * g10) * inv;
        float u01 = (t00 * g01 + t01 * g11) * inv;
        float u10 = (t10 * g00 + t11 * g10) * inv;
        float u11 = (t10 * g01 + t11 * g11) * inv;
        t00 = u00; t01 = u01; t10 = u10; t11 = u11;
        a00 += t00; a01 += t01; a10 += t10; a11 += t11;
    }
    const float tr0 = trg[0], tr1 = trg[1];

    // ---------------- per-lane layer-0 constants: k = hi*16 + i -------------
    float w00[16], w01[16], b0v[16];
    #pragma unroll
    for (int i = 0; i < 16; ++i) {
        int k = hi * 16 + i;
        w00[i] = w0g[k];
        w01[i] = w0g[32 + k];
        b0v[i] = b0g[k];
    }

    // ---------------- A-matrix weights (Wᵀ), f16 hi/lo split -----------------
    // A k order used here (contiguous, same as B): lanes 0-15 hold K in {0..15}, lanes 16-31
    // hold K in {16..31}; v16h element i -> K = hi*16 + i.
    // Row M = p; tile t covers output features t*16 + M.
    // BN scale is folded into A2 rows; BN shift + b2 into layer-2 C-init.
    v16h a1h0, a1l0, a1h1, a1l1;
    v16h a2h0, a2l0, a2h1, a2l1;
    #pragma unroll
    for (int i = 0; i < 16; ++i) {
        int K = hi * 16 + i;
        {
            float w = w1g[K * 32 + p];
            _Float16 h = (_Float16)w;
            a1h0[i] = h; a1l0[i] = (_Float16)(w - (float)h);
        }
        {
            float w = w1g[K * 32 + 16 + p];
            _Float16 h = (_Float16)w;
            a1h1[i] = h; a1l1[i] = (_Float16)(w - (float)h);
        }
        {
            int f = p;
            float s = gmg[f] * rsqrtf(vrg[f] + 1.0e-3f);
            float w = s * w2g[K * 32 + f];
            _Float16 h = (_Float16)w;
            a2h0[i] = h; a2l0[i] = (_Float16)(w - (float)h);
        }
        {
            int f = 16 + p;
            float s = gmg[f] * rsqrtf(vrg[f] + 1.0e-3f);
            float w = s * w2g[K * 32 + f];
            _Float16 h = (_Float16)w;
            a2h1[i] = h; a2l1[i] = (_Float16)(w - (float)h);
        }
    }

    // ---------------- C-matrix inits (bias / fused BN shift) ----------------
    // C/D layout: lane n, VGPR j -> feature M = j + (hi ? 8 : 0) (+16 for tile1)
    v8f c1i0, c1i1, c2i0, c2i1;
    #pragma unroll
    for (int j = 0; j < 8; ++j) {
        int f0 = j + hi * 8;
        int f1 = 16 + j + hi * 8;
        c1i0[j] = b1g[f0];
        c1i1[j] = b1g[f1];
        float s0 = gmg[f0] * rsqrtf(vrg[f0] + 1.0e-3f);
        float s1 = gmg[f1] * rsqrtf(vrg[f1] + 1.0e-3f);
        c2i0[j] = s0 * (b2g[f0] - mng[f0]) + btg[f0];
        c2i1[j] = s1 * (b2g[f1] - mng[f1]) + btg[f1];
    }

    // ---------------- final projection weights (STRENGTH folded in) --------
    // D-layout feature owned by slot i: i<8 -> f = i + hi*8 (tile0),
    //                                   i>=8 -> f = 16 + (i-8) + hi*8 (tile1)
    float wfx[16], wfy[16];
    #pragma unroll
    for (int i = 0; i < 16; ++i) {
        int f = (i < 8) ? (i + hi * 8) : (16 + (i - 8) + hi * 8);
        wfx[i] = 5.0f * wfg[f * 2 + 0];
        wfy[i] = 5.0f * wfg[f * 2 + 1];
    }

    const int wavesPerBlock = blockDim.x >> 5;
    const int wid     = blockIdx.x * wavesPerBlock + (threadIdx.x >> 5);
    const int wstride = gridDim.x * wavesPerBlock;

    for (int g = wid; g < NGROUPS; g += wstride) {
        const int pix = g * 16 + p;

        if (g + wstride < NGROUPS) {
            __builtin_prefetch(&xg[(size_t)(g + wstride) * 48 + p * 3], 0, 1);
        }

        const float cx = xg[pix * 3 + 0];
        const float cy = xg[pix * 3 + 1];
        const float c0 = a00 * cx + a01 * cy + tr0;
        const float c1 = a10 * cx + a11 * cy + tr1;
        const float y0 = c0 * 0.1f, y1 = c1 * 0.1f;

        // ---- layer 0 (exact f32 VALU) straight into B layout: K = hi*16+i --
        v16h bh, bl;
        #pragma unroll
        for (int i = 0; i < 16; ++i) {
            float v = fast_cosf(fmaf(y0, w00[i], fmaf(y1, w01[i], b0v[i])));
            _Float16 h = (_Float16)v;
            bh[i] = h; bl[i] = (_Float16)(v - (float)h);
        }

        // ---- layer 1: D = W1ᵀ·Y + b1, 3-term hi/lo WMMA, 2 tiles -----------
        v8f acc0 = c1i0, acc1 = c1i1;
        acc0 = __builtin_amdgcn_wmma_f32_16x16x32_f16(false, a1l0, false, bh, (short)0, acc0, false, false);
        acc0 = __builtin_amdgcn_wmma_f32_16x16x32_f16(false, a1h0, false, bl, (short)0, acc0, false, false);
        acc0 = __builtin_amdgcn_wmma_f32_16x16x32_f16(false, a1h0, false, bh, (short)0, acc0, false, false);
        acc1 = __builtin_amdgcn_wmma_f32_16x16x32_f16(false, a1l1, false, bh, (short)0, acc1, false, false);
        acc1 = __builtin_amdgcn_wmma_f32_16x16x32_f16(false, a1h1, false, bl, (short)0, acc1, false, false);
        acc1 = __builtin_amdgcn_wmma_f32_16x16x32_f16(false, a1h1, false, bh, (short)0, acc1, false, false);

        // ---- cos, then D layout -> B layout (swap 8 values across xor-16) --
        float yv[16];
        #pragma unroll
        for (int j = 0; j < 8; ++j) {
            yv[j]     = fast_cosf(acc0[j]);
            yv[8 + j] = fast_cosf(acc1[j]);
        }
        #pragma unroll
        for (int j = 0; j < 8; ++j) {
            float send = hi ? yv[j] : yv[8 + j];  // hi lane gives feats 8..15, lo gives 16..23
            float recv = xor16(send);
            float e_lo = hi ? recv : yv[j];       // K offset j
            float e_hi = hi ? yv[8 + j] : recv;   // K offset 8+j
            _Float16 h0 = (_Float16)e_lo; bh[j]     = h0; bl[j]     = (_Float16)(e_lo - (float)h0);
            _Float16 h1 = (_Float16)e_hi; bh[8 + j] = h1; bl[8 + j] = (_Float16)(e_hi - (float)h1);
        }

        // ---- layer 2 (BN scale folded into A2, shift into C-init) ----------
        acc0 = c2i0; acc1 = c2i1;
        acc0 = __builtin_amdgcn_wmma_f32_16x16x32_f16(false, a2l0, false, bh, (short)0, acc0, false, false);
        acc0 = __builtin_amdgcn_wmma_f32_16x16x32_f16(false, a2h0, false, bl, (short)0, acc0, false, false);
        acc0 = __builtin_amdgcn_wmma_f32_16x16x32_f16(false, a2h0, false, bh, (short)0, acc0, false, false);
        acc1 = __builtin_amdgcn_wmma_f32_16x16x32_f16(false, a2l1, false, bh, (short)0, acc1, false, false);
        acc1 = __builtin_amdgcn_wmma_f32_16x16x32_f16(false, a2h1, false, bl, (short)0, acc1, false, false);
        acc1 = __builtin_amdgcn_wmma_f32_16x16x32_f16(false, a2h1, false, bh, (short)0, acc1, false, false);

        #pragma unroll
        for (int j = 0; j < 8; ++j) {
            yv[j]     = fast_cosf(acc0[j]);
            yv[8 + j] = fast_cosf(acc1[j]);
        }

        // ---- final 32->2 projection (partial per half-wave, xor-16 reduce) -
        float px = 0.f, py = 0.f;
        #pragma unroll
        for (int i = 0; i < 16; ++i) {
            px = fmaf(yv[i], wfx[i], px);
            py = fmaf(yv[i], wfy[i], py);
        }
        px += xor16(px);
        py += xor16(py);

        float cr = c0 + px;
        float cc = c1 + py;

        // ---- bilinear sample of (512,512) image ----------------------------
        cr = fminf(fmaxf(cr, 0.0f), 511.0f);
        cc = fminf(fmaxf(cc, 0.0f), 511.0f);
        float fr = floorf(cr), fc = floorf(cc);
        int r0 = (int)fr, q0 = (int)fc;
        int r1 = min(r0 + 1, 511), q1 = min(q0 + 1, 511);
        float wr = cr - fr, wc = cc - fc;
        float v00 = img[r0 * 512 + q0];
        float v01 = img[r0 * 512 + q1];
        float v10 = img[r1 * 512 + q0];
        float v11 = img[r1 * 512 + q1];
        float val = (1.f - wr) * ((1.f - wc) * v00 + wc * v01)
                  +        wr  * ((1.f - wc) * v10 + wc * v11);

        if (hi == 0) out[pix] = val;
    }
}

extern "C" void kernel_launch(void* const* d_in, const int* in_sizes, int n_in,
                              void* d_out, int out_size, void* d_ws, size_t ws_size,
                              hipStream_t stream) {
    (void)in_sizes; (void)n_in; (void)out_size; (void)d_ws; (void)ws_size;
    const float* xg  = (const float*)d_in[0];
    const float* gen = (const float*)d_in[1];
    const float* trg = (const float*)d_in[2];
    const float* w0g = (const float*)d_in[3];
    const float* b0g = (const float*)d_in[4];
    const float* w1g = (const float*)d_in[5];
    const float* b1g = (const float*)d_in[6];
    const float* w2g = (const float*)d_in[7];
    const float* b2g = (const float*)d_in[8];
    const float* gmg = (const float*)d_in[9];
    const float* btg = (const float*)d_in[10];
    const float* mng = (const float*)d_in[11];
    const float* vrg = (const float*)d_in[12];
    const float* wfg = (const float*)d_in[13];
    const float* img = (const float*)d_in[14];
    float* out = (float*)d_out;

    // 1024 blocks x 8 waves = 8192 waves; 65536 groups -> 8 groups/wave
    siren_warp_wmma<<<1024, 256, 0, stream>>>(
        xg, gen, trg, w0g, b0g, w1g, b1g, w2g, b2g,
        gmg, btg, mng, vrg, wfg, img, out);
}
